// QAttentionModel_80135499808774
// MI455X (gfx1250) — matrix operand order repaired and checked by reading, not compiled, not run on hardware
//
#include <hip/hip_runtime.h>

// Problem constants (from reference): B=8, S=2048, E=256, DK=64, H=64
#define SLEN  2048
#define BATCH 8
#define EMB   256
#define DKV   64
#define HID   64
#define MROWS (BATCH * SLEN)   // 16384

typedef __bf16 bf16;
typedef __attribute__((ext_vector_type(8)))  __bf16 bf16x8;
typedef __attribute__((ext_vector_type(16))) __bf16 bf16x16;
typedef __attribute__((ext_vector_type(8)))  float  v8f;

// ---------------------------------------------------------------------------
// WMMA operand builders (CDNA5 wave32 layouts, ISA 7.12.2), bf16-native:
// every build is two 16-byte vector loads per lane, no conversions.
//
// A (16x32, MxK): lane<16 -> row m=lane, elems {K 0..7, K 16..23}
//                 lane>=16 -> row m=lane-16, elems {K 8..15, K 24..31}
// B (32x16, KxN): lane n holds K {kadd..kadd+7, kadd+16..kadd+23} (kadd = 0 | 8)
//                 -> source must be [N][K] with K contiguous ("row-major in N")
// C/D (16x16 f32): elem r, lane l -> m = r + 8*(l>>4), n = l&15
// ---------------------------------------------------------------------------
static __device__ inline bf16x16 join16(bf16x8 lo, bf16x8 hi) {
  return __builtin_shufflevector(lo, hi, 0, 1, 2, 3, 4, 5, 6, 7,
                                 8, 9, 10, 11, 12, 13, 14, 15);
}

static __device__ inline bf16x16 load_a_bf(const bf16* src, int rowstride, int kbase) {
  const int lane = threadIdx.x & 31;
  const int m    = lane & 15;
  const int klo  = (lane >> 4) << 3;           // 0 or 8
  const bf16* p  = src + (size_t)m * rowstride + kbase + klo;
  return join16(*(const bf16x8*)p, *(const bf16x8*)(p + 16));
}

static __device__ inline bf16x16 load_b_bf(const bf16* src, int rowstride) {
  const int lane = threadIdx.x & 31;
  const int n    = lane & 15;
  const int kadd = (lane >> 4) << 3;           // 0 or 8
  const bf16* p  = src + (size_t)n * rowstride + kadd;
  return join16(*(const bf16x8*)p, *(const bf16x8*)(p + 16));
}

static __device__ inline v8f wmma_bf16(bf16x16 a, bf16x16 b, v8f c) {
  return __builtin_amdgcn_wmma_f32_16x16x32_bf16(false, a, false, b, (short)0, c,
                                                 false, false);
}

// ---------------------------------------------------------------------------
// Prep kernel 1: obs fp32 -> bf16 (one-time, coalesced)
// ---------------------------------------------------------------------------
__global__ void cvt_obs_kernel(const float* __restrict__ obs,
                               bf16* __restrict__ obs_bf) {
  const size_t i = (size_t)blockIdx.x * blockDim.x + threadIdx.x;
  if (i < (size_t)MROWS * EMB) obs_bf[i] = (bf16)obs[i];
}

// ---------------------------------------------------------------------------
// Prep kernel 2: transpose+convert weights into B-operand-friendly layouts.
//   wt[w][dk][e]  = W{q,k,v}[e][dk]   (3 x 64 x 256)
//   w1t[h][dk]    = W1[dk][h]         (64 x 64)
// ---------------------------------------------------------------------------
__global__ void prep_weights_kernel(const float* __restrict__ Wq,
                                    const float* __restrict__ Wk,
                                    const float* __restrict__ Wv,
                                    const float* __restrict__ W1,
                                    bf16* __restrict__ wt,
                                    bf16* __restrict__ w1t) {
  const int nth = gridDim.x * blockDim.x;
  const int tid = blockIdx.x * blockDim.x + threadIdx.x;
  for (int i = tid; i < 3 * DKV * EMB; i += nth) {
    const int w = i / (DKV * EMB);
    const int r = i - w * (DKV * EMB);
    const int dk = r / EMB;
    const int e  = r - dk * EMB;
    const float* W = (w == 0) ? Wq : (w == 1) ? Wk : Wv;
    wt[i] = (bf16)W[e * DKV + dk];
  }
  for (int i = tid; i < HID * DKV; i += nth) {
    const int h  = i / DKV;
    const int dk = i - h * DKV;
    w1t[i] = (bf16)W1[dk * HID + h];
  }
}

// ---------------------------------------------------------------------------
// Kernel: QKV projection.  grid (M/16, 3), block 32 (one wave).
// q,k scaled by DK^-0.25 and stored bf16 row-major [row][dk];
// v stored bf16 TRANSPOSED: vt[b][dk][s]  (B-operand-ready for P@V).
// ---------------------------------------------------------------------------
__global__ void qkv_proj_kernel(const bf16* __restrict__ obs_bf,
                                const bf16* __restrict__ wt,
                                bf16* __restrict__ qbf,
                                bf16* __restrict__ kbf,
                                bf16* __restrict__ vt) {
  const int tile  = blockIdx.x;
  const int which = blockIdx.y;                // 0=q, 1=k, 2=v
  const bf16* W = wt + (size_t)which * DKV * EMB;
  const int row0 = tile * 16;

  v8f acc[4] = {v8f{0}, v8f{0}, v8f{0}, v8f{0}};
  for (int kc = 0; kc < EMB; kc += 32) {
    bf16x16 a = load_a_bf(obs_bf + (size_t)row0 * EMB, EMB, kc);
#pragma unroll
    for (int nt = 0; nt < 4; ++nt) {
      bf16x16 b = load_b_bf(W + (size_t)(nt * 16) * EMB + kc, EMB);
      acc[nt] = wmma_bf16(a, b, acc[nt]);
    }
  }

  const int lane = threadIdx.x & 31;
  const int mh   = (lane >> 4) << 3;
  const int nn   = lane & 15;
  if (which < 2) {
    const float fac = 0.35355339059327373f;    // 64^-0.25
    bf16* outb = (which == 0) ? qbf : kbf;
#pragma unroll
    for (int nt = 0; nt < 4; ++nt)
#pragma unroll
      for (int r = 0; r < 8; ++r)
        outb[(size_t)(row0 + r + mh) * DKV + nt * 16 + nn] = (bf16)(acc[nt][r] * fac);
  } else {
    const int bb = row0 / SLEN;                // tile never crosses batch (2048%16==0)
    const int s0 = row0 - bb * SLEN;
#pragma unroll
    for (int nt = 0; nt < 4; ++nt)
#pragma unroll
      for (int r = 0; r < 8; ++r)
        vt[((size_t)bb * DKV + nt * 16 + nn) * SLEN + s0 + r + mh] = (bf16)acc[nt][r];
  }
}

// ---------------------------------------------------------------------------
// Kernel: fused causal attention + softmax + attn write + P@V + MLP.
// grid (S/16, B), block 128 (4 waves).  Dynamic LDS (~70KB -> 4 WGs/WGP):
//   sc[16*2048] bf16 (64KB) | obuf[16*64] bf16 | hbuf[16*64] f32
// ---------------------------------------------------------------------------
__global__ void attn_fused_kernel(const bf16* __restrict__ qbf,
                                  const bf16* __restrict__ kbf,
                                  const bf16* __restrict__ vt,
                                  const bf16* __restrict__ w1t,
                                  const float* __restrict__ bias1,
                                  const float* __restrict__ W2,
                                  const float* __restrict__ bias2,
                                  float* __restrict__ out) {
  extern __shared__ char smem[];
  bf16*  sc   = (bf16*)smem;                                // 16 x SLEN bf16
  bf16*  obuf = (bf16*)(smem + 16 * SLEN * 2);              // 16 x 64 bf16
  float* hbuf = (float*)(smem + 16 * SLEN * 2 + 16 * DKV * 2); // 16 x 64 f32

  const int qt = blockIdx.x;
  const int b  = blockIdx.y;
  const int q0 = qt * 16;
  const int ncols = (qt + 1) * 16;
  const int tid  = threadIdx.x;
  const int w    = tid >> 5;
  const int lane = tid & 31;
  const int mrow = (lane >> 4) << 3;
  const int ncol = lane & 15;

  const bf16* Q = qbf + ((size_t)b * SLEN + q0) * DKV;
  const bf16* K = kbf + (size_t)b * SLEN * DKV;
  const bf16* V = vt + (size_t)b * DKV * SLEN;

  // ---- pass 1: scores = q @ k^T, causal mask, into LDS (waves stripe kt) ----
  bf16x16 a0 = load_a_bf(Q, DKV, 0);
  bf16x16 a1 = load_a_bf(Q, DKV, 32);
  for (int kt = w; kt <= qt; kt += 4) {
    v8f c = {0};
    bf16x16 kb0 = load_b_bf(K + (size_t)kt * 16 * DKV, DKV);
    bf16x16 kb1 = load_b_bf(K + (size_t)kt * 16 * DKV + 32, DKV);
    c = wmma_bf16(a0, kb0, c);
    c = wmma_bf16(a1, kb1, c);
    const int key = kt * 16 + ncol;
#pragma unroll
    for (int r = 0; r < 8; ++r) {
      const int m = r + mrow;
      sc[m * SLEN + key] = (key <= q0 + m) ? (bf16)c[r] : (bf16)(-__builtin_inff());
    }
  }
  __syncthreads();

  // ---- pass 2: row softmax in LDS (8 lanes per row, shuffle reduce) ----
  {
    const int row = tid >> 3;
    const int part = tid & 7;
    bf16* sr = sc + row * SLEN;
    float mx = -__builtin_inff();
    for (int c = part; c < ncols; c += 8) mx = fmaxf(mx, (float)sr[c]);
#pragma unroll
    for (int s = 4; s >= 1; s >>= 1) mx = fmaxf(mx, __shfl_xor(mx, s, 8));
    float sum = 0.0f;
    for (int c = part; c < ncols; c += 8) {
      float e = __expf((float)sr[c] - mx);
      sr[c] = (bf16)e;
      sum += e;
    }
#pragma unroll
    for (int s = 4; s >= 1; s >>= 1) sum += __shfl_xor(sum, s, 8);
    const float inv = 1.0f / sum;
    for (int c = part; c < ncols; c += 8) sr[c] = (bf16)((float)sr[c] * inv);
  }
  __syncthreads();

  // ---- pass 3: write attn (coalesced, zeros above diagonal) ----
  {
    float* attn = out + (size_t)BATCH * SLEN
                + ((size_t)b * SLEN + q0) * (size_t)SLEN;
    for (int idx = tid; idx < 16 * SLEN; idx += 128) {
      const int c = idx & (SLEN - 1);
      attn[idx] = (c < ncols) ? (float)sc[idx] : 0.0f;
    }
  }

  // ---- pass 4: O = P @ V, each wave owns one 16-wide dk tile ----
  v8f oacc = {0};
  {
    const int klo = (lane >> 4) << 3;
    for (int kc = 0; kc < ncols; kc += 32) {
      const bool full = (kc + 32 <= ncols);
      const bf16* pr = sc + (lane & 15) * SLEN + kc + klo;
      bf16x8 lo = *(const bf16x8*)pr;
      bf16x8 hi;
      if (full) hi = *(const bf16x8*)(pr + 16);
      else      hi = bf16x8{};
      bf16x16 a  = join16(lo, hi);
      bf16x16 vb = load_b_bf(V + (size_t)(w * 16) * SLEN + kc, SLEN);
      oacc = wmma_bf16(a, vb, oacc);
    }
  }
#pragma unroll
  for (int r = 0; r < 8; ++r)
    obuf[(r + mrow) * DKV + w * 16 + ncol] = (bf16)oacc[r];
  __syncthreads();

  // ---- pass 5: h = relu(O @ W1 + b1), each wave one 16-wide H tile ----
  v8f hacc = {0};
  for (int kc = 0; kc < DKV; kc += 32) {
    bf16x16 a  = load_a_bf(obuf, DKV, kc);
    bf16x16 wb = load_b_bf(w1t + (size_t)(w * 16) * DKV + kc, DKV);
    hacc = wmma_bf16(a, wb, hacc);
  }
  {
    const int col = w * 16 + ncol;
    const float bb = bias1[col];
#pragma unroll
    for (int r = 0; r < 8; ++r)
      hbuf[(r + mrow) * HID + col] = fmaxf(hacc[r] + bb, 0.0f);
  }
  __syncthreads();

  // ---- pass 6: x = h @ W2 + b2 (per-row dot, shuffle reduce) ----
  {
    const int row = tid >> 3;
    const int part = tid & 7;
    float acc = 0.0f;
    for (int j = part; j < HID; j += 8) acc += hbuf[row * HID + j] * W2[j];
#pragma unroll
    for (int s = 4; s >= 1; s >>= 1) acc += __shfl_xor(acc, s, 8);
    if (part == 0)
      out[(size_t)b * SLEN + q0 + row] = acc + bias2[0];
  }
}

// ---------------------------------------------------------------------------
extern "C" void kernel_launch(void* const* d_in, const int* in_sizes, int n_in,
                              void* d_out, int out_size, void* d_ws, size_t ws_size,
                              hipStream_t stream) {
  const float* obs = (const float*)d_in[0];
  // d_in[1] = act (unused by forward)
  const float* Wq  = (const float*)d_in[2];
  const float* Wk  = (const float*)d_in[3];
  const float* Wv  = (const float*)d_in[4];
  const float* W1  = (const float*)d_in[5];
  const float* b1  = (const float*)d_in[6];
  const float* W2  = (const float*)d_in[7];
  const float* b2  = (const float*)d_in[8];
  float* out = (float*)d_out;

  // bf16 workspace layout (~14.1 MB total)
  bf16* obs_bf = (bf16*)d_ws;                          // M*E
  bf16* qbf    = obs_bf + (size_t)MROWS * EMB;         // M*64
  bf16* kbf    = qbf + (size_t)MROWS * DKV;            // M*64
  bf16* vt     = kbf + (size_t)MROWS * DKV;            // B*64*S
  bf16* wt     = vt + (size_t)MROWS * DKV;             // 3*64*256
  bf16* w1t    = wt + (size_t)3 * DKV * EMB;           // 64*64

  cvt_obs_kernel<<<(MROWS * EMB + 255) / 256, 256, 0, stream>>>(obs, obs_bf);
  prep_weights_kernel<<<64, 256, 0, stream>>>(Wq, Wk, Wv, W1, wt, w1t);

  dim3 g1(MROWS / 16, 3);
  qkv_proj_kernel<<<g1, 32, 0, stream>>>(obs_bf, wt, qbf, kbf, vt);

  dim3 g2(SLEN / 16, BATCH);
  const size_t lds_bytes = (size_t)16 * SLEN * 2 + (size_t)16 * DKV * 2
                         + (size_t)16 * HID * 4;
  attn_fused_kernel<<<g2, 128, lds_bytes, stream>>>(qbf, kbf, vt, w1t, b1, W2, b2, out);
}
